// DiscretizedManifoldBlock_33139967656629
// MI455X (gfx1250) — compile-verified, matrix operand order checked by reading, not run on hardware
//
#include <hip/hip_runtime.h>
#include <hip/hip_bf16.h>
#include <stdint.h>

// ---------------------------------------------------------------------------
// Types for CDNA5 WMMA (wave32, 16x16x32 bf16 -> f32)
// ---------------------------------------------------------------------------
typedef __attribute__((ext_vector_type(16))) __bf16 v16bf;
typedef __attribute__((ext_vector_type(4)))  __bf16 v4bf;
typedef __attribute__((ext_vector_type(8)))  float  v8f;

// Problem constants (from reference)
constexpr int    kB   = 4;
constexpr int    kT   = 4096;
constexpr int    kC   = 1024;
constexpr int    kH   = 16;
constexpr int    kD   = 64;
constexpr int    kNC  = 16;    // chunks = T / 256
constexpr int    kCH  = 256;   // CHUNK
constexpr int    kLEVELS = 4;
constexpr int    kCODES  = 1024;
constexpr size_t kS      = (size_t)kB * kT * kC;     // 16,777,216
constexpr int    kRowsBT   = kB * kT;                // 16384
constexpr int    kRowsComb = kB * kH * kNC * kCH;    // 262144

// ---------------------------------------------------------------------------
// WMMA GEMM:  Co[M,N] = act( scale * (A[M,K] . W[N,K]^T) + bias[N] )
// act: 0=none, 1=sigmoid, 2=feg(alpha)   (alpha read from device scalar)
// Block: 256 threads (8 wave32), tile BM=128 x BN=128, BK=32, double-buffered
// LDS in bf16. Each wave computes a 32(M) x 64(N) block = 2x4 WMMA tiles.
// ---------------------------------------------------------------------------
#define BM 128
#define BN 128
#define BK 32

__global__ __launch_bounds__(256)
void k_gemm(const float* __restrict__ A, const float* __restrict__ W,
            const float* __restrict__ bias, float* __restrict__ Co,
            int M, int N, int K, float scale, int act,
            const float* __restrict__ alphap)
{
  __shared__ __align__(16) __bf16 tA[2][BM * BK];  // [m][k]
  __shared__ __align__(16) __bf16 tB[2][BN * BK];  // [n][k]

  const int tid  = threadIdx.x;
  const int lane = tid & 31;
  const int wave = tid >> 5;    // 0..7
  const int wm   = wave >> 1;   // 0..3  (M direction, 32 rows each)
  const int wn   = wave & 1;    // 0..1  (N direction, 64 cols each)
  const int lr   = lane & 15;
  const int kh   = lane >> 4;   // K-half selector per ISA 16-bit layout
  const int bM   = blockIdx.y * BM;
  const int bN   = blockIdx.x * BN;

  const v8f zacc = {0.f, 0.f, 0.f, 0.f, 0.f, 0.f, 0.f, 0.f};
  v8f acc[2][4];
#pragma unroll
  for (int i = 0; i < 2; ++i)
#pragma unroll
    for (int j = 0; j < 4; ++j) acc[i][j] = zacc;

  auto stage = [&](int kpos, int bi) {
    // prefetch two K-steps ahead (emits global_prefetch_b8)
    if (kpos + 2 * BK < K) {
      int prow = tid & 127;
      __builtin_prefetch(A + (size_t)(bM + prow) * K + kpos + 2 * BK, 0, 1);
      int pn = bN + prow;
      if (pn < N) __builtin_prefetch(W + (size_t)pn * K + kpos + 2 * BK, 0, 1);
    }
#pragma unroll
    for (int i = 0; i < 4; ++i) {
      int slot = tid + i * 256;        // 0..1023 float4 slots per tile
      int row  = slot >> 3;            // 8 float4 per 32-wide K row
      int kk   = (slot & 7) << 2;
      float4 va = *(const float4*)(A + (size_t)(bM + row) * K + kpos + kk);
      v4bf pa;                         // native f32->bf16 (v_cvt_pk_bf16_f32)
      pa.x = (__bf16)va.x; pa.y = (__bf16)va.y;
      pa.z = (__bf16)va.z; pa.w = (__bf16)va.w;
      *(v4bf*)(&tA[bi][row * BK + kk]) = pa;

      int n = bN + row;
      float4 vb = make_float4(0.f, 0.f, 0.f, 0.f);
      if (n < N) vb = *(const float4*)(W + (size_t)n * K + kpos + kk);
      v4bf pb;
      pb.x = (__bf16)vb.x; pb.y = (__bf16)vb.y;
      pb.z = (__bf16)vb.z; pb.w = (__bf16)vb.w;
      *(v4bf*)(&tB[bi][row * BK + kk]) = pb;
    }
  };

  union FragU { v16bf v; uint4 q[2]; };
  auto frag = [&](const __bf16* tile, int row) -> v16bf {
    // 16-bit operand layout: elems 0..7 -> K = kh*8 + 0..7,
    //                        elems 8..15 -> K = 16 + kh*8 + 0..7
    FragU f;
    f.q[0] = *(const uint4*)(tile + row * BK + kh * 8);
    f.q[1] = *(const uint4*)(tile + row * BK + 16 + kh * 8);
    return f.v;
  };

  const int nk = K / BK;
  stage(0, 0);
  __syncthreads();
  for (int kt = 0; kt < nk; ++kt) {
    if (kt + 1 < nk) stage((kt + 1) * BK, (kt + 1) & 1);
    const __bf16* cA = tA[kt & 1];
    const __bf16* cB = tB[kt & 1];
    v16bf af[2], bfr[4];
#pragma unroll
    for (int sm = 0; sm < 2; ++sm) af[sm] = frag(cA, wm * 32 + sm * 16 + lr);
#pragma unroll
    for (int sn = 0; sn < 4; ++sn) bfr[sn] = frag(cB, wn * 64 + sn * 16 + lr);
#pragma unroll
    for (int sm = 0; sm < 2; ++sm)
#pragma unroll
      for (int sn = 0; sn < 4; ++sn)
        acc[sm][sn] = __builtin_amdgcn_wmma_f32_16x16x32_bf16(
            false, af[sm], false, bfr[sn], (short)0, acc[sm][sn], false, false);
    __syncthreads();
  }

  const float alpha = alphap ? *alphap : 0.f;
#pragma unroll
  for (int sn = 0; sn < 4; ++sn) {
    int n = bN + wn * 64 + sn * 16 + lr;
    if (n >= N) continue;
    float bv = bias ? bias[n] : 0.f;
#pragma unroll
    for (int sm = 0; sm < 2; ++sm) {
      int mrow = bM + wm * 32 + sm * 16 + kh * 8;
#pragma unroll
      for (int r = 0; r < 8; ++r) {
        float v = scale * acc[sm][sn][r] + bv;
        if (act == 1)      v = 1.f / (1.f + __expf(-v));
        else if (act == 2) v = v + alpha * v * __expf(-0.5f * v * v);
        Co[(size_t)(mrow + r) * N + n] = v;
      }
    }
  }
}

// ---------------------------------------------------------------------------
// Generic LayerNorm over last dim (cols), one row per block.
//   y[row] = LN(x[row] (+ x2[row])) * g + b  (+ res[row])
// ---------------------------------------------------------------------------
__global__ __launch_bounds__(256)
void k_ln(const float* __restrict__ x, const float* __restrict__ x2,
          const float* __restrict__ g, const float* __restrict__ bp,
          const float* __restrict__ res, float* __restrict__ y, int cols)
{
  __shared__ float r1[256];
  __shared__ float r2[256];
  const size_t row = blockIdx.x;
  const int tid = threadIdx.x;
  const float* xr  = x + row * cols;
  const float* x2r = x2 ? x2 + row * cols : nullptr;

  float s = 0.f, s2 = 0.f;
  for (int c = tid; c < cols; c += 256) {
    float v = xr[c];
    if (x2r) v += x2r[c];
    s += v; s2 += v * v;
  }
  r1[tid] = s; r2[tid] = s2;
  __syncthreads();
  for (int off = 128; off; off >>= 1) {
    if (tid < off) { r1[tid] += r1[tid + off]; r2[tid] += r2[tid + off]; }
    __syncthreads();
  }
  const float inv  = 1.f / (float)cols;
  const float mean = r1[0] * inv;
  const float var  = r2[0] * inv - mean * mean;
  const float rstd = rsqrtf(var + 1e-5f);
  for (int c = tid; c < cols; c += 256) {
    float v = xr[c];
    if (x2r) v += x2r[c];
    float o = (v - mean) * rstd * g[c] + bp[c];
    if (res) o += res[row * cols + c];
    y[row * cols + c] = o;
  }
}

// ---------------------------------------------------------------------------
// Elementwise add: y = a + b (size kS)
// ---------------------------------------------------------------------------
__global__ __launch_bounds__(256)
void k_add(const float* __restrict__ a, const float* __restrict__ b,
           float* __restrict__ y)
{
  size_t i = (size_t)blockIdx.x * 256 + threadIdx.x;
  y[i] = a[i] + b[i];
}

// ---------------------------------------------------------------------------
// chunk sums: cs[b,h,c,d] = sum_t gates*marks over the chunk
// grid = B*H*NC blocks, 64 threads (d)
// ---------------------------------------------------------------------------
__global__ __launch_bounds__(64)
void k_chunk_sums(const float* __restrict__ marks, const float* __restrict__ gates,
                  float* __restrict__ cs)
{
  const int blk = blockIdx.x;              // ((b*H + h)*NC + c)
  const int c = blk % kNC;
  const int h = (blk / kNC) % kH;
  const int b = blk / (kNC * kH);
  const int d = threadIdx.x;
  float s = 0.f;
  for (int t = 0; t < kCH; ++t) {
    size_t gi = ((size_t)b * kT + (size_t)c * kCH + t) * kC + h * kD + d;
    s += gates[gi] * marks[gi];
  }
  cs[(size_t)blk * kD + d] = s;
}

// exclusive prefix over chunks: carry[b,h,c,d]
// grid = B*H blocks, 64 threads
__global__ __launch_bounds__(64)
void k_carry(const float* __restrict__ cs, float* __restrict__ carry)
{
  const int bh = blockIdx.x;
  const int d = threadIdx.x;
  float run = 0.f;
  for (int c = 0; c < kNC; ++c) {
    size_t i = ((size_t)bh * kNC + c) * kD + d;
    carry[i] = run;
    run += cs[i];
  }
}

// ---------------------------------------------------------------------------
// Build comb[row][128]: [0:64) = xh, [64:128) = LN_card(cards_local)
// Phase 1 (64 d-threads): serial gm-cumsum over t; write cards_local to global.
// Phase 2 (256 t-threads): LN over d of card half + copy xh.
// grid = B*H*NC blocks, 256 threads.
// ---------------------------------------------------------------------------
__global__ __launch_bounds__(256)
void k_build_comb(const float* __restrict__ marks, const float* __restrict__ gates,
                  const float* __restrict__ ncarry, const float* __restrict__ xln,
                  const float* __restrict__ card_g, const float* __restrict__ card_b,
                  float* __restrict__ comb)
{
  const int blk = blockIdx.x;              // ((b*H + h)*NC + c)
  const int c = blk % kNC;
  const int h = (blk / kNC) % kH;
  const int b = blk / (kNC * kH);
  const int tid = threadIdx.x;
  const size_t rowbase = (size_t)blk * kCH;

  if (tid < kD) {
    const int d = tid;
    const float ncd = ncarry[(size_t)blk * kD + d];
    float run = 0.f;
    for (int t = 0; t < kCH; ++t) {
      // cards_local[t] = ncarry + (t==0 ? 0 : lc[t-1])
      comb[(rowbase + t) * (2 * kD) + kD + d] = ncd + run;
      size_t gi = ((size_t)b * kT + (size_t)c * kCH + t) * kC + h * kD + d;
      run += gates[gi] * marks[gi];
    }
  }
  __threadfence_block();
  __syncthreads();

  const int t = tid;
  float* rowp = comb + (rowbase + t) * (2 * kD);
  float s = 0.f, s2 = 0.f;
  for (int d = 0; d < kD; ++d) { float v = rowp[kD + d]; s += v; s2 += v * v; }
  const float mean = s * (1.f / kD);
  const float var  = s2 * (1.f / kD) - mean * mean;
  const float rstd = rsqrtf(var + 1e-5f);
  for (int d = 0; d < kD; ++d) {
    float v = rowp[kD + d];
    rowp[kD + d] = (v - mean) * rstd * card_g[d] + card_b[d];
    rowp[d] = xln[((size_t)b * kT + (size_t)c * kCH + t) * kC + h * kD + d];
  }
}

// ---------------------------------------------------------------------------
// Scatter head output back to token layout:
// out[b, c*CH+t, h*D+d] = ho[(((b*H+h)*NC)+c)*CH + t][d]
// ---------------------------------------------------------------------------
__global__ __launch_bounds__(256)
void k_scatter(const float* __restrict__ ho, float* __restrict__ out)
{
  size_t i = (size_t)blockIdx.x * 256 + threadIdx.x;   // over kS
  int ch = (int)(i % kC);
  size_t tg = (i / kC) % kT;
  int b = (int)(i / ((size_t)kC * kT));
  int h = ch >> 6, d = ch & 63;
  int c = (int)(tg >> 8), t = (int)(tg & 255);
  size_t row = (((size_t)b * kH + h) * kNC + c) * kCH + t;
  out[i] = ho[row * kD + d];
}

// ---------------------------------------------------------------------------
// Codebook squared norms: cbn[l*CODES+n] = sum_k cb[l][n][k]^2
// grid = LEVELS*CODES blocks, 256 threads
// ---------------------------------------------------------------------------
__global__ __launch_bounds__(256)
void k_codebook_norm(const float* __restrict__ cb, float* __restrict__ cbn)
{
  __shared__ float red[256];
  const size_t rn = blockIdx.x;
  const int tid = threadIdx.x;
  const float* row = cb + rn * kC;
  float s = 0.f;
  for (int c = tid; c < kC; c += 256) { float v = row[c]; s += v * v; }
  red[tid] = s;
  __syncthreads();
  for (int off = 128; off; off >>= 1) {
    if (tid < off) red[tid] += red[tid + off];
    __syncthreads();
  }
  if (tid == 0) cbn[rn] = red[0];
}

// ---------------------------------------------------------------------------
// Row argmin over CODES scores (scores = -2*dot + ||cb||^2; row norm is const)
// grid = rows, 256 threads. Writes int index + float index to d_out section.
// ---------------------------------------------------------------------------
__global__ __launch_bounds__(256)
void k_argmin(const float* __restrict__ sc, int* __restrict__ idx,
              float* __restrict__ idxf)
{
  __shared__ float bv[256];
  __shared__ int   bix[256];
  const size_t row = blockIdx.x;
  const int tid = threadIdx.x;
  float best = 3.4e38f;
  int   bi   = 0;
  for (int n = tid; n < kCODES; n += 256) {
    float v = sc[row * kCODES + n];
    if (v < best) { best = v; bi = n; }
  }
  bv[tid] = best; bix[tid] = bi;
  __syncthreads();
  for (int off = 128; off; off >>= 1) {
    if (tid < off) {
      float v2 = bv[tid + off]; int i2 = bix[tid + off];
      if (v2 < bv[tid] || (v2 == bv[tid] && i2 < bix[tid])) {
        bv[tid] = v2; bix[tid] = i2;
      }
    }
    __syncthreads();
  }
  if (tid == 0) { idx[row] = bix[0]; idxf[row] = (float)bix[0]; }
}

// ---------------------------------------------------------------------------
// RVQ apply: q = cb[idx]; loss += (q - residual)^2; total += q; residual -= q
// grid = rows, 256 threads
// ---------------------------------------------------------------------------
__global__ __launch_bounds__(256)
void k_rvq_apply(const float* __restrict__ cbl, const int* __restrict__ idx,
                 float* __restrict__ total, float* __restrict__ resid,
                 float* __restrict__ acc)
{
  __shared__ float red[256];
  const size_t row = blockIdx.x;
  const int tid = threadIdx.x;
  const int id = idx[row];
  const float* q = cbl + (size_t)id * kC;
  float p = 0.f;
  for (int c = tid; c < kC; c += 256) {
    size_t i = row * kC + c;
    float qq = q[c];
    float r = resid[i];
    float d = qq - r;
    p += d * d;
    total[i] += qq;
    resid[i] = r - qq;
  }
  red[tid] = p;
  __syncthreads();
  for (int off = 128; off; off >>= 1) {
    if (tid < off) red[tid] += red[tid + off];
    __syncthreads();
  }
  if (tid == 0) atomicAdd(acc, red[0]);
}

// ---------------------------------------------------------------------------
// Finalize: out = xcur + total; write losses
// ---------------------------------------------------------------------------
__global__ __launch_bounds__(256)
void k_finalize(const float* __restrict__ xcur, const float* __restrict__ total,
                const float* __restrict__ acc, float* __restrict__ dout)
{
  size_t i = (size_t)blockIdx.x * 256 + threadIdx.x;
  dout[i] = xcur[i] + total[i];
  if (i == 0) {
    float L = acc[0] / ((float)kS * (float)kLEVELS);
    dout[kS]     = L;   // q_loss
    dout[kS + 1] = L;   // e_loss (numerically identical)
  }
}

// ---------------------------------------------------------------------------
// Host orchestration
// ---------------------------------------------------------------------------
extern "C" void kernel_launch(void* const* d_in, const int* in_sizes, int n_in,
                              void* d_out, int out_size, void* d_ws, size_t ws_size,
                              hipStream_t stream)
{
  const float* x       = (const float*)d_in[0];
  const float* ln1_g   = (const float*)d_in[1];
  const float* ln1_b   = (const float*)d_in[2];
  const float* ln2_g   = (const float*)d_in[3];
  const float* ln2_b   = (const float*)d_in[4];
  const float* ln3_g   = (const float*)d_in[5];
  const float* ln3_b   = (const float*)d_in[6];
  const float* Wm      = (const float*)d_in[7];
  const float* bm      = (const float*)d_in[8];
  const float* Wg      = (const float*)d_in[9];
  const float* bg      = (const float*)d_in[10];
  const float* carry_g = (const float*)d_in[11];
  const float* carry_b = (const float*)d_in[12];
  const float* card_g  = (const float*)d_in[13];
  const float* card_b  = (const float*)d_in[14];
  const float* hW1     = (const float*)d_in[15];
  const float* hb1     = (const float*)d_in[16];
  const float* h_alpha = (const float*)d_in[17];
  const float* hW2     = (const float*)d_in[18];
  const float* hb2     = (const float*)d_in[19];
  const float* Wp      = (const float*)d_in[20];
  const float* bp      = (const float*)d_in[21];
  const float* aln_g   = (const float*)d_in[22];
  const float* aln_b   = (const float*)d_in[23];
  const float* mW1     = (const float*)d_in[24];
  const float* mb1     = (const float*)d_in[25];
  const float* m_alpha = (const float*)d_in[26];
  const float* mW2     = (const float*)d_in[27];
  const float* mb2     = (const float*)d_in[28];
  const float* cb      = (const float*)d_in[29];
  float* dout = (float*)d_out;

  float* ws = (float*)d_ws;
  float* B0 = ws;              // S : xln / LN scratch
  float* B1 = ws + 1 * kS;     // S : marks / ho / gemm out / scores
  float* B2 = ws + 2 * kS;     // S : gates / scatter out / residual
  float* Bx = ws + 3 * kS;     // S : running x
  float* Ba = ws + 4 * kS;     // S : attention output a
  float* B3 = ws + 5 * kS;     // 2S : comb  (MLP intermediate spans B3..B4 = 4S)
  float* B4 = ws + 7 * kS;     // 2S : h     (later: RVQ total uses first S)
  float* SMp   = ws + 9 * kS;  // small region
  float* cs    = SMp;                  // 65536
  float* carry = SMp + 65536;          // 65536
  float* ncar  = SMp + 131072;         // 65536
  float* cbn   = SMp + 196608;         // 4096
  int*   idxb  = (int*)(SMp + 200704); // 16384 ints
  float* acc   = SMp + 217088;         // 2 floats

  auto gemm = [&](const float* A, const float* Wt, const float* bias, float* Co,
                  int M, int N, int K, float scale, int act, const float* al) {
    dim3 g((N + BN - 1) / BN, M / BM);
    k_gemm<<<g, 256, 0, stream>>>(A, Wt, bias, Co, M, N, K, scale, act, al);
  };
  auto ln = [&](const float* xi, const float* x2, const float* g, const float* b,
                const float* res, float* y, int rows, int cols) {
    k_ln<<<rows, 256, 0, stream>>>(xi, x2, g, b, res, y, cols);
  };

  auto attn = [&](const float* xin, float* aout) {
    gemm(xin, Wm, bm, B1, kRowsBT, kC, kC, 1.f, 0, nullptr);          // marks
    gemm(xin, Wg, bg, B2, kRowsBT, kC, kC, 1.f, 1, nullptr);          // gates
    k_chunk_sums<<<kB * kH * kNC, 64, 0, stream>>>(B1, B2, cs);
    k_carry<<<kB * kH, 64, 0, stream>>>(cs, carry);
    ln(carry, nullptr, carry_g, carry_b, nullptr, ncar, kB * kH * kNC, kD);
    k_build_comb<<<kB * kH * kNC, 256, 0, stream>>>(B1, B2, ncar, xin,
                                                    card_g, card_b, B3);
    gemm(B3, hW1, hb1, B4, kRowsComb, 2 * kD, 2 * kD, 1.f, 2, h_alpha); // h (feg)
    gemm(B4, hW2, hb2, B1, kRowsComb, kD, 2 * kD, 1.f, 0, nullptr);     // ho
    k_scatter<<<(int)(kS / 256), 256, 0, stream>>>(B1, B2);
    gemm(B2, Wp, bp, B1, kRowsBT, kC, kC, 1.f, 0, nullptr);             // proj
    ln(B1, nullptr, aln_g, aln_b, xin, aout, kRowsBT, kC);              // xin+LN
  };

  // a = attn(LN(x))
  ln(x, nullptr, ln1_g, ln1_b, nullptr, B0, kRowsBT, kC);
  attn(B0, Ba);
  // a = attn(LN(x + a))   (DELIB = 2)
  ln(x, Ba, ln1_g, ln1_b, nullptr, B0, kRowsBT, kC);
  attn(B0, Ba);
  // xcur = x + a
  k_add<<<(int)(kS / 256), 256, 0, stream>>>(x, Ba, Bx);

  // MLP
  ln(Bx, nullptr, ln2_g, ln2_b, nullptr, B0, kRowsBT, kC);
  gemm(B0, mW1, mb1, B3, kRowsBT, 4 * kC, kC, 1.f, 2, m_alpha);   // [16384,4096]
  gemm(B3, mW2, mb2, B1, kRowsBT, kC, 4 * kC, 1.f, 0, nullptr);
  k_add<<<(int)(kS / 256), 256, 0, stream>>>(Bx, B1, Bx);

  // RVQ
  ln(Bx, nullptr, ln3_g, ln3_b, nullptr, B0, kRowsBT, kC);        // xn
  hipMemcpyAsync(B2, B0, kS * sizeof(float), hipMemcpyDeviceToDevice, stream);
  hipMemsetAsync(B4, 0, kS * sizeof(float), stream);              // total
  hipMemsetAsync(acc, 0, 2 * sizeof(float), stream);
  k_codebook_norm<<<kLEVELS * kCODES, 256, 0, stream>>>(cb, cbn);
  for (int l = 0; l < kLEVELS; ++l) {
    const float* cbl = cb + (size_t)l * kCODES * kC;
    gemm(B2, cbl, cbn + (size_t)l * kCODES, B1, kRowsBT, kCODES, kC,
         -2.f, 0, nullptr);                                       // dist-ish
    k_argmin<<<kRowsBT, 256, 0, stream>>>(B1, idxb,
        dout + kS + 2 + (size_t)l * kRowsBT);
    k_rvq_apply<<<kRowsBT, 256, 0, stream>>>(cbl, idxb, B4, B2, acc);
  }
  k_finalize<<<(int)(kS / 256), 256, 0, stream>>>(Bx, B4, acc, dout);
}
